// MOEBlock_84920093376645
// MI455X (gfx1250) — compile-verified, matrix operand order checked by reading, not run on hardware
//
#include <hip/hip_runtime.h>
#include <math.h>

typedef __bf16 bf16;
typedef __attribute__((ext_vector_type(16))) __bf16 v16bf;
typedef __attribute__((ext_vector_type(8)))  __bf16 v8bf;
typedef __attribute__((ext_vector_type(8)))  float  v8f;

// ---- problem constants ----
#define BB   8
#define NN   1024
#define CC   768
#define HH   12
#define HD   64
#define TT   (BB*NN)          // 8192 tokens
#define EE   23
#define KK3  3
#define HID  576
#define C3   (3*CC)           // 2304
#define OUTN ((long)TT*CC)    // 6291456 floats of main output

// ---- GEMM tiling: 128x64 block tile, 8 waves each computing 32x32 (2x2 WMMA) ----
#define TM 128
#define TN 64
#define KC 32
#define AS_STRIDE 40   // bf16 elems; 80B row stride (16B aligned)
#define BS_STRIDE 40

__device__ __forceinline__ v16bf frag16(const bf16* p) {
  v8bf lo = *reinterpret_cast<const v8bf*>(p);
  v8bf hi = *reinterpret_cast<const v8bf*>(p + 16);
  v16bf r;
#pragma unroll
  for (int j = 0; j < 8; ++j) { r[j] = lo[j]; r[j + 8] = hi[j]; }
  return r;
}

__device__ __forceinline__ float gelu_exact(float v) {
  return 0.5f * v * (1.0f + erff(v * 0.70710678118654752f));
}

// Async 16B global->LDS copy (CDNA5 ASYNCcnt path). INST_OFFSET is added to
// both the LDS and global addresses per CDNA5 ISA.
__device__ __forceinline__ void async_copy16(unsigned ldsaddr, const bf16* gsrc) {
  asm volatile("global_load_async_to_lds_b128 %0, %1, off"
               :: "v"(ldsaddr), "v"(gsrc) : "memory");
}
__device__ __forceinline__ void async_copy16_off16(unsigned ldsaddr, const bf16* gsrc) {
  asm volatile("global_load_async_to_lds_b128 %0, %1, off offset:16"
               :: "v"(ldsaddr), "v"(gsrc) : "memory");
}
__device__ __forceinline__ void wait_async0() {
  asm volatile("s_wait_asynccnt 0x0" ::: "memory");
}

// Generic bf16-A x fp32-B (converted to bf16 on LDS fill) tile GEMM.
// Double-buffered LDS, one barrier per K-step, 4 WMMAs per wave per K-step.
// A tile is filled with async global->LDS copies (no VGPR round trip);
// B tile converts fp32->bf16 through VGPRs.
// rowmap(r_local in [0,128)) -> global A row.  epi(r_local, global_col, value).
template <typename RowMap, typename Epi>
__device__ __forceinline__ void gemm_core(const bf16* __restrict__ A, int lda,
                                          const float* __restrict__ Bw, int ldb,
                                          int K, int n0, RowMap rowmap, Epi epi) {
  __shared__ bf16 As[2][TM * AS_STRIDE];
  __shared__ bf16 Bs[2][TN * BS_STRIDE];
  const int tid  = threadIdx.x;
  const int lane = tid & 31;
  const int wave = tid >> 5;
  const int wm   = wave >> 1;           // 0..3 -> 32-row band
  const int wn   = wave & 1;            // 0..1 -> 32-col band
  const int l15  = lane & 15;
  const int kb   = (lane < 16) ? 0 : 8; // fragment K split per ISA layout

  // A fill: 128 rows x 32 cols bf16; thread: row=tid>>1, 16 elems at (tid&1)*16
  const int  arow = tid >> 1;
  const int  aseg = (tid & 1) * 16;
  const long ag   = rowmap(arow);
  const bf16* aptr = A + ag * lda + aseg;
  const unsigned ldsA0 =
      (unsigned)(uintptr_t)(&As[0][arow * AS_STRIDE + aseg]);
  const unsigned ldsA1 =
      (unsigned)(uintptr_t)(&As[1][arow * AS_STRIDE + aseg]);
  // B fill: 32(K) x 64(N) fp32 -> bf16; thread: col n = tid&63, 8 K's at (tid>>6)*8
  const int bn  = tid & 63;
  const int bkg = (tid >> 6) * 8;
  const float* bptr = Bw + (long)bkg * ldb + n0 + bn;

  auto load_tile = [&](int k0, int buf) {
    const bf16* gsrc = aptr + k0;
    const unsigned ldsa = buf ? ldsA1 : ldsA0;
    async_copy16(ldsa, gsrc);          // bytes [0,16)
    async_copy16_off16(ldsa, gsrc);    // bytes [16,32)
    alignas(16) bf16 tmp[8];
#pragma unroll
    for (int j = 0; j < 8; ++j) tmp[j] = (bf16)bptr[(long)(k0 + j) * ldb];
    *reinterpret_cast<uint4*>(&Bs[buf][bn * BS_STRIDE + bkg]) =
        *reinterpret_cast<const uint4*>(tmp);
  };

  v8f acc[2][2] = {};
  int cur = 0;
  load_tile(0, 0);
  for (int k0 = 0; k0 < K; k0 += KC) {
    wait_async0();
    __syncthreads();
    if (k0 + KC < K) load_tile(k0 + KC, cur ^ 1);
    v16bf af[2], bfr[2];
#pragma unroll
    for (int t = 0; t < 2; ++t) {
      af[t]  = frag16(&As[cur][(wm * 32 + t * 16 + l15) * AS_STRIDE + kb]);
      bfr[t] = frag16(&Bs[cur][(wn * 32 + t * 16 + l15) * BS_STRIDE + kb]);
    }
#pragma unroll
    for (int i = 0; i < 2; ++i)
#pragma unroll
      for (int j = 0; j < 2; ++j)
        acc[i][j] = __builtin_amdgcn_wmma_f32_16x16x32_bf16(
            false, af[i], false, bfr[j], (short)0, acc[i][j], false, false);
    cur ^= 1;
  }
  const int rsub = (lane & 16) ? 8 : 0;
#pragma unroll
  for (int i = 0; i < 2; ++i) {
    const int rb = wm * 32 + i * 16 + rsub;
#pragma unroll
    for (int j = 0; j < 2; ++j) {
      const int col = n0 + wn * 32 + j * 16 + l15;
#pragma unroll
      for (int q = 0; q < 8; ++q) epi(rb + q, col, acc[i][j][q]);
    }
  }
}

// ---------------- elementwise / small kernels ----------------

__global__ __launch_bounds__(256) void k_ln(const float* __restrict__ x,
                                            const float* __restrict__ g,
                                            const float* __restrict__ b,
                                            bf16* __restrict__ out) {
  __shared__ float r1[256], r2[256];
  const long t = blockIdx.x;
  const float* xr = x + t * CC;
  float s = 0.f, s2 = 0.f;
  for (int i = threadIdx.x; i < CC; i += 256) { float v = xr[i]; s += v; s2 += v * v; }
  r1[threadIdx.x] = s; r2[threadIdx.x] = s2;
  __syncthreads();
  for (int st = 128; st > 0; st >>= 1) {
    if (threadIdx.x < st) { r1[threadIdx.x] += r1[threadIdx.x + st]; r2[threadIdx.x] += r2[threadIdx.x + st]; }
    __syncthreads();
  }
  const float mean = r1[0] * (1.0f / CC);
  const float var  = r2[0] * (1.0f / CC) - mean * mean;
  const float rs   = rsqrtf(var + 1e-5f);
  for (int i = threadIdx.x; i < CC; i += 256)
    out[t * CC + i] = (bf16)((xr[i] - mean) * rs * g[i] + b[i]);
}

__global__ __launch_bounds__(128) void k_zero_small(int* p) {
  p[threadIdx.x] = 0;   // clears cnt/offsets/cursor/psum block (128 slots)
}

__global__ __launch_bounds__(256) void k_gate(const bf16* __restrict__ h2,
                                              const float* __restrict__ wg,
                                              const float* __restrict__ bg,
                                              const float* __restrict__ mbias,
                                              float* __restrict__ gw,
                                              int* __restrict__ top_i,
                                              float* __restrict__ top_w,
                                              int* __restrict__ cnt,
                                              float* __restrict__ psum) {
  __shared__ float xt[CC];
  __shared__ float gws[EE];
  __shared__ float ssum;
  const long t = blockIdx.x;
  const int tid = threadIdx.x;
  for (int i = tid; i < CC; i += 256) xt[i] = (float)h2[t * CC + i];
  __syncthreads();
  if (tid < EE) {
    float s = 0.f;
    for (int c = 0; c < CC; ++c) s += xt[c] * wg[(long)c * EE + tid];
    s += bg[tid];
    float z = 1.0f / (1.0f + __expf(-s));
    gws[tid] = z;
    gw[t * EE + tid] = z;
  }
  __syncthreads();
  if (tid == 0) {
    float sg = 0.f;
    for (int e = 0; e < EE; ++e) sg += gws[e];
    ssum = sg;
    bool used[EE];
    for (int e = 0; e < EE; ++e) used[e] = false;
    int sel[KK3]; float selw[KK3]; float wsum = 0.f;
    for (int k = 0; k < KK3; ++k) {
      float best = -1e30f; int bi = 0;
      for (int e = 0; e < EE; ++e)
        if (!used[e]) { float sc = gws[e] + mbias[e]; if (sc > best) { best = sc; bi = e; } }
      used[bi] = true; sel[k] = bi; selw[k] = gws[bi]; wsum += gws[bi];
    }
    for (int k = 0; k < KK3; ++k) {
      top_i[t * KK3 + k] = sel[k];
      top_w[t * KK3 + k] = selw[k] / wsum;
      atomicAdd(&cnt[sel[k]], 1);
    }
  }
  __syncthreads();
  if (tid < EE) atomicAdd(&psum[tid], gws[tid] / ssum);
}

__global__ __launch_bounds__(32) void k_scan_aux(const int* __restrict__ cnt,
                                                 int* __restrict__ offv,
                                                 int* __restrict__ cursor,
                                                 const float* __restrict__ psum,
                                                 float* __restrict__ outf) {
  const int tid = threadIdx.x;
  if (tid == 0) {
    int o = 0;
    for (int e = 0; e < EE; ++e) { offv[e] = o; o += cnt[e]; }
    offv[EE] = o;
  }
  __syncthreads();
  if (tid < EE) cursor[tid] = offv[tid];
  if (tid == 0) {
    float aux = 0.f;
    for (int e = 0; e < EE; ++e) {
      float P = psum[e] / (float)TT;
      float F = (float)EE * (float)cnt[e] / (float)(KK3 * TT);
      aux += P * F;
    }
    outf[OUTN] = aux;
  }
  if (tid < EE) reinterpret_cast<int*>(outf)[OUTN + 1 + tid] = cnt[tid];
}

__global__ __launch_bounds__(256) void k_scatter(const int* __restrict__ top_i,
                                                 const float* __restrict__ top_w,
                                                 int* __restrict__ cursor,
                                                 int* __restrict__ ptok,
                                                 float* __restrict__ pw) {
  const int t = blockIdx.x * 256 + threadIdx.x;
  for (int k = 0; k < KK3; ++k) {
    int e = top_i[t * KK3 + k];
    int pos = atomicAdd(&cursor[e], 1);
    ptok[pos] = t;
    pw[pos] = top_w[t * KK3 + k];
  }
}

__global__ __launch_bounds__(256) void k_init_acc(const float* __restrict__ sb2,
                                                  const float* __restrict__ eb2,
                                                  const int* __restrict__ top_i,
                                                  const float* __restrict__ top_w,
                                                  float* __restrict__ acc) {
  const long t = blockIdx.x;
  const int c = blockIdx.y * 256 + threadIdx.x;
  float v = sb2[c];
  for (int k = 0; k < KK3; ++k) {
    int e = top_i[t * KK3 + k];
    v += top_w[t * KK3 + k] * eb2[(long)e * CC + c];
  }
  acc[t * CC + c] = v;
}

__global__ __launch_bounds__(256) void k_final(const float* __restrict__ x1,
                                               const float* __restrict__ acc,
                                               float* __restrict__ outf) {
  const long i = (long)blockIdx.x * 256 + threadIdx.x;
  outf[i] = x1[i] + acc[i];
}

// ---------------- WMMA GEMM kernels ----------------

__global__ __launch_bounds__(256) void k_gemm_qkv(const bf16* __restrict__ A,
                                                  const float* __restrict__ Bw,
                                                  bf16* __restrict__ qb,
                                                  bf16* __restrict__ kb,
                                                  bf16* __restrict__ vb) {
  const int m0 = blockIdx.x * TM, n0 = blockIdx.y * TN;
  gemm_core(A, CC, Bw, C3, CC, n0,
      [=](int r) -> long { return m0 + r; },
      [=](int r, int col, float v) {
        int t = m0 + r;
        int s = col / CC, rem = col - s * CC;
        int h = rem >> 6, d = rem & 63;
        int b = t >> 10, n = t & 1023;
        long idx = (((long)(b * HH + h)) * NN + n) * HD + d;
        bf16 val = (bf16)v;
        if (s == 0) qb[idx] = val; else if (s == 1) kb[idx] = val; else vb[idx] = val;
      });
}

__global__ __launch_bounds__(256) void k_gemm_proj(const bf16* __restrict__ A,
                                                   const float* __restrict__ Bw,
                                                   const float* __restrict__ bproj,
                                                   const float* __restrict__ xin,
                                                   float* __restrict__ x1) {
  const int m0 = blockIdx.x * TM, n0 = blockIdx.y * TN;
  gemm_core(A, CC, Bw, CC, CC, n0,
      [=](int r) -> long { return m0 + r; },
      [=](int r, int col, float v) {
        long i = (long)(m0 + r) * CC + col;
        x1[i] = xin[i] + v + bproj[col];
      });
}

__global__ __launch_bounds__(256) void k_gemm_s1(const bf16* __restrict__ A,
                                                 const float* __restrict__ Bw,
                                                 const float* __restrict__ sb1,
                                                 bf16* __restrict__ sh) {
  const int m0 = blockIdx.x * TM, n0 = blockIdx.y * TN;
  gemm_core(A, CC, Bw, HID, CC, n0,
      [=](int r) -> long { return m0 + r; },
      [=](int r, int col, float v) {
        sh[(long)(m0 + r) * HID + col] = (bf16)gelu_exact(v + sb1[col]);
      });
}

__global__ __launch_bounds__(256) void k_gemm_s2(const bf16* __restrict__ A,
                                                 const float* __restrict__ Bw,
                                                 float* __restrict__ acc) {
  const int m0 = blockIdx.x * TM, n0 = blockIdx.y * TN;
  gemm_core(A, HID, Bw, CC, HID, n0,
      [=](int r) -> long { return m0 + r; },
      [=](int r, int col, float v) {
        acc[(long)(m0 + r) * CC + col] += v;
      });
}

__global__ __launch_bounds__(256) void k_gemm_e1(const bf16* __restrict__ h2,
                                                 const float* __restrict__ eW1,
                                                 const float* __restrict__ eb1,
                                                 const int* __restrict__ offv,
                                                 const int* __restrict__ ptok,
                                                 bf16* __restrict__ hmid) {
  const int e = blockIdx.z;
  const int beg = offv[e], end = offv[e + 1];
  const int base = beg + blockIdx.x * TM;
  if (base >= end) return;
  const int n0 = blockIdx.y * TN;
  const float* Bw = eW1 + (long)e * CC * HID;
  gemm_core(h2, CC, Bw, HID, CC, n0,
      [=](int r) -> long { int gr = base + r; if (gr >= end) gr = beg; return (long)ptok[gr]; },
      [=](int r, int col, float v) {
        int gr = base + r;
        if (gr < end)
          hmid[(long)gr * HID + col] = (bf16)gelu_exact(v + eb1[(long)e * HID + col]);
      });
}

__global__ __launch_bounds__(256) void k_gemm_e2(const bf16* __restrict__ hmid,
                                                 const float* __restrict__ eW2,
                                                 const int* __restrict__ offv,
                                                 const int* __restrict__ ptok,
                                                 const float* __restrict__ pw,
                                                 float* __restrict__ acc) {
  const int e = blockIdx.z;
  const int beg = offv[e], end = offv[e + 1];
  const int base = beg + blockIdx.x * TM;
  if (base >= end) return;
  const int n0 = blockIdx.y * TN;
  const float* Bw = eW2 + (long)e * HID * CC;
  gemm_core(hmid, HID, Bw, CC, HID, n0,
      [=](int r) -> long { int gr = base + r; if (gr >= end) gr = beg; return (long)gr; },
      [=](int r, int col, float v) {
        int gr = base + r;
        if (gr < end) {
          int t = ptok[gr];
          atomicAdd(&acc[(long)t * CC + col], pw[gr] * v);
        }
      });
}

// ---------------- flash attention (Br=32, Bc=64, hd=64) ----------------
#define AD 72
__global__ __launch_bounds__(256) void k_attn(const bf16* __restrict__ qb,
                                              const bf16* __restrict__ kb,
                                              const bf16* __restrict__ vb,
                                              bf16* __restrict__ ab) {
  __shared__ bf16 Qs[32 * AD];
  __shared__ bf16 Ks[64 * AD];
  __shared__ bf16 Vts[64 * AD];
  __shared__ bf16 Ps[32 * AD];
  __shared__ float Ss[32 * 64];
  __shared__ float Os[32 * 64];
  __shared__ float ms[32], ls[32], as_[32];

  const int tid = threadIdx.x, lane = tid & 31, wave = tid >> 5;
  const int wm = wave >> 2, wn = wave & 3;   // 2 x 4 waves over 32x64
  const int l15 = lane & 15;
  const int kb_ = (lane < 16) ? 0 : 8;

  const int bh = blockIdx.x;
  const int b = bh / HH, h = bh % HH;
  const long base = (long)bh * NN;
  const int q0 = blockIdx.y * 32;

  { int row = tid >> 3, seg = (tid & 7) * 8;
    *reinterpret_cast<uint4*>(&Qs[row * AD + seg]) =
        *reinterpret_cast<const uint4*>(qb + (base + q0 + row) * HD + seg); }
  for (int i = tid; i < 32 * 64; i += 256) Os[i] = 0.f;
  if (tid < 32) { ms[tid] = -1e30f; ls[tid] = 0.f; }

  for (int c0 = 0; c0 < NN; c0 += 64) {
    __syncthreads();
#pragma unroll
    for (int it = 0; it < 2; ++it) {
      int task = tid + it * 256;
      int row = task >> 3, seg = (task & 7) * 8;
      *reinterpret_cast<uint4*>(&Ks[row * AD + seg]) =
          *reinterpret_cast<const uint4*>(kb + (base + c0 + row) * HD + seg);
      uint4 vv = *reinterpret_cast<const uint4*>(vb + (base + c0 + row) * HD + seg);
      const bf16* vp = reinterpret_cast<const bf16*>(&vv);
#pragma unroll
      for (int j = 0; j < 8; ++j) Vts[(seg + j) * AD + row] = vp[j];
    }
    __syncthreads();
    // S = Q K^T
    v8f s = {};
#pragma unroll
    for (int kk = 0; kk < 64; kk += 32) {
      v16bf aq = frag16(&Qs[(wm * 16 + l15) * AD + kk + kb_]);
      v16bf bk2 = frag16(&Ks[(wn * 16 + l15) * AD + kk + kb_]);
      s = __builtin_amdgcn_wmma_f32_16x16x32_bf16(false, aq, false, bk2,
                                                  (short)0, s, false, false);
    }
    { int colc = wn * 16 + l15;
      int rb = wm * 16 + ((lane & 16) ? 8 : 0);
#pragma unroll
      for (int i = 0; i < 8; ++i) Ss[(rb + i) * 64 + colc] = s[i] * 0.125f; }
    __syncthreads();
    if (tid < 32) {
      int r = tid;
      float mo = ms[r], mn = mo;
      for (int j = 0; j < 64; ++j) mn = fmaxf(mn, Ss[r * 64 + j]);
      float al = __expf(mo - mn);
      float ln_ = ls[r] * al;
      for (int j = 0; j < 64; ++j) {
        float p = __expf(Ss[r * 64 + j] - mn);
        ln_ += p;
        Ps[r * AD + j] = (bf16)p;
      }
      ms[r] = mn; ls[r] = ln_; as_[r] = al;
    }
    __syncthreads();
    for (int i = tid; i < 32 * 64; i += 256) Os[i] *= as_[i >> 6];
    __syncthreads();
    // O += P V
    v8f o = {};
#pragma unroll
    for (int kk = 0; kk < 64; kk += 32) {
      v16bf ap = frag16(&Ps[(wm * 16 + l15) * AD + kk + kb_]);
      v16bf bv = frag16(&Vts[(wn * 16 + l15) * AD + kk + kb_]);
      o = __builtin_amdgcn_wmma_f32_16x16x32_bf16(false, ap, false, bv,
                                                  (short)0, o, false, false);
    }
    { int colc = wn * 16 + l15;
      int rb = wm * 16 + ((lane & 16) ? 8 : 0);
#pragma unroll
      for (int i = 0; i < 8; ++i) Os[(rb + i) * 64 + colc] += o[i]; }
  }
  __syncthreads();
  for (int i = tid; i < 32 * 64; i += 256) {
    int r = i >> 6, d = i & 63;
    float v = Os[i] / ls[r];
    ab[((long)(b * NN + q0 + r)) * CC + h * HD + d] = (bf16)v;
  }
}

// ---------------- launcher ----------------

extern "C" void kernel_launch(void* const* d_in, const int* in_sizes, int n_in,
                              void* d_out, int out_size, void* d_ws, size_t ws_size,
                              hipStream_t stream) {
  const float* x      = (const float*)d_in[0];
  const float* ln1_g  = (const float*)d_in[1];
  const float* ln1_b  = (const float*)d_in[2];
  const float* w_qkv  = (const float*)d_in[3];
  const float* w_proj = (const float*)d_in[4];
  const float* b_proj = (const float*)d_in[5];
  const float* ln2_g  = (const float*)d_in[6];
  const float* ln2_b  = (const float*)d_in[7];
  const float* w_gate = (const float*)d_in[8];
  const float* b_gate = (const float*)d_in[9];
  const float* mbias  = (const float*)d_in[10];
  const float* eW1    = (const float*)d_in[11];
  const float* eb1    = (const float*)d_in[12];
  const float* eW2    = (const float*)d_in[13];
  const float* eb2    = (const float*)d_in[14];
  const float* sW1    = (const float*)d_in[15];
  const float* sb1    = (const float*)d_in[16];
  const float* sW2    = (const float*)d_in[17];
  const float* sb2    = (const float*)d_in[18];
  float* outf = (float*)d_out;

  char* ws = (char*)d_ws;
  size_t o = 0;
  auto take = [&](size_t bytes) -> char* {
    char* p = ws + o;
    o = (o + bytes + 255) & ~(size_t)255;
    return p;
  };
  bf16* qkvbuf = (bf16*)take((size_t)3 * TT * CC * 2);  // q|k|v ; hmid overlays
  bf16* qb = qkvbuf;
  bf16* kb = qkvbuf + (size_t)TT * CC;
  bf16* vb = qkvbuf + (size_t)2 * TT * CC;
  bf16* hmid = qkvbuf;                                  // 3T*HID*2 <= 3T*C*2
  bf16* ab = (bf16*)take((size_t)TT * CC * 2);
  bf16* shb = ab;                                       // T*HID*2 <= T*C*2
  bf16* hb = (bf16*)take((size_t)TT * CC * 2);          // LN1 out, reused for LN2
  float* x1  = (float*)take((size_t)TT * CC * 4);
  float* acc = (float*)take((size_t)TT * CC * 4);
  float* gw  = (float*)take((size_t)TT * EE * 4);
  int*   top_i = (int*)take((size_t)TT * KK3 * 4);
  float* top_w = (float*)take((size_t)TT * KK3 * 4);
  int*   ptok  = (int*)take((size_t)KK3 * TT * 4);
  float* pw    = (float*)take((size_t)KK3 * TT * 4);
  int*   smal  = (int*)take(512);
  int*   cnt    = smal;
  int*   offv   = smal + 32;
  int*   cursor = smal + 64;
  float* psum   = (float*)(smal + 96);

  k_zero_small<<<1, 128, 0, stream>>>(smal);
  k_ln<<<TT, 256, 0, stream>>>(x, ln1_g, ln1_b, hb);
  k_gemm_qkv<<<dim3(TT / TM, C3 / TN), 256, 0, stream>>>(hb, w_qkv, qb, kb, vb);
  k_attn<<<dim3(BB * HH, NN / 32), 256, 0, stream>>>(qb, kb, vb, ab);
  k_gemm_proj<<<dim3(TT / TM, CC / TN), 256, 0, stream>>>(ab, w_proj, b_proj, x, x1);
  k_ln<<<TT, 256, 0, stream>>>(x1, ln2_g, ln2_b, hb);
  k_gate<<<TT, 256, 0, stream>>>(hb, w_gate, b_gate, mbias, gw, top_i, top_w, cnt, psum);
  k_scan_aux<<<1, 32, 0, stream>>>(cnt, offv, cursor, psum, outf);
  k_scatter<<<TT / 256, 256, 0, stream>>>(top_i, top_w, cursor, ptok, pw);
  k_init_acc<<<dim3(TT, CC / 256), 256, 0, stream>>>(sb2, eb2, top_i, top_w, acc);
  k_gemm_s1<<<dim3(TT / TM, HID / TN), 256, 0, stream>>>(hb, sW1, sb1, shb);
  k_gemm_s2<<<dim3(TT / TM, CC / TN), 256, 0, stream>>>(shb, sW2, acc);
  k_gemm_e1<<<dim3((KK3 * TT) / TM, HID / TN, EE), 256, 0, stream>>>(hb, eW1, eb1, offv, ptok, hmid);
  k_gemm_e2<<<dim3((KK3 * TT) / TM, CC / TN, EE), 256, 0, stream>>>(hmid, eW2, offv, ptok, pw, acc);
  k_final<<<(int)(OUTN / 256), 256, 0, stream>>>(x1, acc, outf);
}
